// Attention_21388937134632
// MI455X (gfx1250) — matrix operand order repaired and checked by reading, not compiled, not run on hardware
//
#include <hip/hip_runtime.h>

#define BATCH 4
#define SEQ   2048
#define DM    1024
#define NH    16
#define DKV   64
#define NBUCK 32
#define RELSPAN (2*SEQ - 1)   /* 4095 */
#define MROWS (BATCH*SEQ)     /* 8192 */

typedef __attribute__((ext_vector_type(16))) __bf16       v16bf;
typedef __attribute__((ext_vector_type(8)))  float        v8f;
typedef __attribute__((ext_vector_type(4)))  float        v4f;
typedef __attribute__((ext_vector_type(8)))  unsigned int v8u;

static __device__ __forceinline__ unsigned short f2bf(float f) {
  unsigned int u = __builtin_bit_cast(unsigned int, f);
  u += 0x7fffu + ((u >> 16) & 1u);          // round-to-nearest-even
  return (unsigned short)(u >> 16);
}

static __device__ __forceinline__ v16bf pack_bf16(uint4 lo, uint4 hi) {
  v8u u = {lo.x, lo.y, lo.z, lo.w, hi.x, hi.y, hi.z, hi.w};
  return __builtin_bit_cast(v16bf, u);
}

// ---------------------------------------------------------------- fp32 -> bf16
__global__ void cvt_bf16_kernel(const float* __restrict__ src,
                                unsigned short* __restrict__ dst, int n4) {
  int i = blockIdx.x * blockDim.x + threadIdx.x;
  if (i >= n4) return;
  float4 f = ((const float4*)src)[i];
  uint2 p;
  p.x = (unsigned)f2bf(f.x) | ((unsigned)f2bf(f.y) << 16);
  p.y = (unsigned)f2bf(f.z) | ((unsigned)f2bf(f.w) << 16);
  ((uint2*)dst)[i] = p;
}

// ---------------------------------------------------------------- T5 bucket LUT
static __device__ __forceinline__ int rel_bucket(int rel) {
  int b  = (rel > 0) ? (NBUCK / 2) : 0;       // 16
  int ar = rel < 0 ? -rel : rel;
  int v;
  if (ar < 8) {
    v = ar;
  } else {
    v = 8 + (int)(__logf((float)ar * 0.125f) * (8.0f / __logf(16.0f)));
    v = v > 15 ? 15 : v;
  }
  return b + v;
}

__global__ void bias_lut_kernel(const float* __restrict__ rel_emb,
                                float* __restrict__ bias_full) {
  int i = blockIdx.x * blockDim.x + threadIdx.x;
  if (i >= NH * RELSPAN) return;
  int h = i / RELSPAN;
  int d = i - h * RELSPAN;
  int bkt = rel_bucket(d - (SEQ - 1));
  bias_full[i] = rel_emb[bkt * NH + h];
}

// ---------------------------------------------------------------- masks output
// 1.07 GB write-once stream: use non-temporal stores so it does not evict the
// L2-resident amask / Q / K / V working set.
__global__ void masks_kernel(const float* __restrict__ bias_full,
                             const float* __restrict__ amask,
                             float* __restrict__ masks) {
  long long row = blockIdx.x;                 // b*NH*SEQ + h*SEQ + q
  int q = (int)(row % SEQ);
  int h = (int)((row / SEQ) % NH);
  long long b = row / (SEQ * NH);
  const float* bsrc = bias_full + h * RELSPAN + (SEQ - 1 - q);
  const v4f*   am4  = (const v4f*)(amask + (b * SEQ + q) * (long long)SEQ);
  v4f*         out4 = (v4f*)(masks + row * SEQ);
  for (int k4 = threadIdx.x; k4 < SEQ / 4; k4 += blockDim.x) {
    int k = 4 * k4;
    v4f a = am4[k4];
    v4f o;
    o.x = a.x + bsrc[k];
    o.y = a.y + bsrc[k + 1];
    o.z = a.z + bsrc[k + 2];
    o.w = a.w + bsrc[k + 3];
    __builtin_nontemporal_store(o, out4 + k4);
  }
}

// ---------------------------------------------------------------- blocked WMMA core
// Each wave computes a 32x64 output tile: 2 M-frags x 4 N-frags, 8 WMMA / K-step.
static __device__ __forceinline__ void gemm_tile_2x4(
    const unsigned short* __restrict__ Abase,   // row (row0+l16), row-major KxDM
    const unsigned short* __restrict__ Bbase,   // row (col0+l16), row-major
    int lh, v8f acc[2][4]) {
  for (int kk = 0; kk < DM; kk += 32) {
    v16bf afrag[2];
#pragma unroll
    for (int mt = 0; mt < 2; ++mt) {
      const unsigned short* ar = Abase + mt * 16 * DM + kk + lh * 8;
      afrag[mt] = pack_bf16(*(const uint4*)ar, *(const uint4*)(ar + 16));
    }
#pragma unroll
    for (int nt = 0; nt < 4; ++nt) {
      const unsigned short* br = Bbase + nt * 16 * DM + kk + lh * 8;
      v16bf bfrag = pack_bf16(*(const uint4*)br, *(const uint4*)(br + 16));
#pragma unroll
      for (int mt = 0; mt < 2; ++mt)
        acc[mt][nt] = __builtin_amdgcn_wmma_f32_16x16x32_bf16(
            false, afrag[mt], false, bfrag, (short)0, acc[mt][nt], false, false);
    }
  }
}

// ---------------------------------------------------------------- QKV GEMM (WMMA)
// Y[r,c] = dot(X[r,:], W[c,:]) + bias[c].
// Q,K stored bf16 [B,H,S,dk]; V stored transposed bf16 [B,H,dk,S] so the
// flash kernel's P*V B-fragments are contiguous b128 global loads.
__global__ void __launch_bounds__(32)
gemm_qkv_kernel(const unsigned short* __restrict__ Xb,
                const unsigned short* __restrict__ Wqb,
                const unsigned short* __restrict__ Wkb,
                const unsigned short* __restrict__ Wvb,
                const float* __restrict__ bq, const float* __restrict__ bk,
                const float* __restrict__ bv,
                unsigned short* __restrict__ Qo,
                unsigned short* __restrict__ Ko,
                unsigned short* __restrict__ Vo) {
  const unsigned short* W; const float* bias; unsigned short* out;
  if (blockIdx.z == 0)      { W = Wqb; bias = bq; out = Qo; }
  else if (blockIdx.z == 1) { W = Wkb; bias = bk; out = Ko; }
  else                      { W = Wvb; bias = bv; out = Vo; }
  const bool vtrans = (blockIdx.z == 2);

  int lane = threadIdx.x;
  int lh   = lane >> 4;
  int l16  = lane & 15;
  int row0 = blockIdx.x * 32;
  int col0 = blockIdx.y * 64;

  v8f acc[2][4];
#pragma unroll
  for (int mt = 0; mt < 2; ++mt)
#pragma unroll
    for (int nt = 0; nt < 4; ++nt)
      acc[mt][nt] = (v8f){0.f,0.f,0.f,0.f,0.f,0.f,0.f,0.f};

  gemm_tile_2x4(Xb + (long long)(row0 + l16) * DM,
                W  + (long long)(col0 + l16) * DM, lh, acc);

  int mbase = lh * 8;
#pragma unroll
  for (int nt = 0; nt < 4; ++nt) {
    int c = col0 + nt * 16 + l16;
    float bcol = bias[c];
    int h = c >> 6, d = c & 63;
#pragma unroll
    for (int mt = 0; mt < 2; ++mt) {
#pragma unroll
      for (int v = 0; v < 8; ++v) {
        int r  = row0 + mt * 16 + mbase + v;
        int bb = r >> 11;                  // r / SEQ
        int s  = r & (SEQ - 1);
        unsigned short val = f2bf(acc[mt][nt][v] + bcol);
        if (vtrans)
          out[(((long long)bb * NH + h) * DKV + d) * SEQ + s] = val;
        else
          out[(((long long)bb * NH + h) * SEQ + s) * DKV + d] = val;
      }
    }
  }
}

// ---------------------------------------------------------------- flash attention
// One wave per 16-row q-tile; keys in steps of 32.  QK^T: 4 WMMA; P*V: 4 WMMA
// straight from the pre-transposed V; P relayout (C-layout -> A-fragment) via a
// private per-wave LDS tile, ordered with s_wait_dscnt (in-order DS pipe).
__global__ void __launch_bounds__(128)
flash_attn_kernel(const unsigned short* __restrict__ Qb,
                  const unsigned short* __restrict__ Kb,
                  const unsigned short* __restrict__ Vtg,   // [B,H,dk,S]
                  const float* __restrict__ bias_full,
                  const float* __restrict__ amask,
                  unsigned short* __restrict__ ctx) {
  __shared__ unsigned short Pl[4][16 * 32];      // per-wave P tile [m][k]

  int tid  = threadIdx.x;
  int wid  = tid >> 5;
  int lane = tid & 31;
  int lh   = lane >> 4;
  int l16  = lane & 15;
  int h = blockIdx.y;
  int b = blockIdx.z;
  int q0 = (blockIdx.x * 4 + wid) * 16;

  const unsigned short* Qt = Qb  + (((long long)b * NH + h) * SEQ) * DKV;
  const unsigned short* Kt = Kb  + (((long long)b * NH + h) * SEQ) * DKV;
  const unsigned short* Vt = Vtg + (((long long)b * NH + h) * DKV) * SEQ;
  const float* brow = bias_full + h * RELSPAN + (SEQ - 1);
  const float* am   = amask + (long long)b * SEQ * SEQ;
  const float  inv  = 1.0f / (float)(DKV * DKV);   // 1/4096

  // Q A-fragments for the 16x64 tile (two K=32 chunks)
  const unsigned short* Qrow = Qt + (long long)(q0 + l16) * DKV;
  v16bf aq[2];
#pragma unroll
  for (int kk = 0; kk < 2; ++kk) {
    uint4 lo = *(const uint4*)(Qrow + kk * 32 + lh * 8);
    uint4 hi = *(const uint4*)(Qrow + kk * 32 + 16 + lh * 8);
    aq[kk] = pack_bf16(lo, hi);
  }

  v8f o[4];
#pragma unroll
  for (int c = 0; c < 4; ++c) o[c] = (v8f){0.f,0.f,0.f,0.f,0.f,0.f,0.f,0.f};
  float rmax[8], rsum[8];
#pragma unroll
  for (int v = 0; v < 8; ++v) { rmax[v] = -1e30f; rsum[v] = 0.f; }

  unsigned short* pl = &Pl[wid][0];

  for (int j0 = 0; j0 < SEQ; j0 += 32) {
    // ---- scores for two 16-key tiles: S = Q(16x64) * K_tile^T(64x16)
    v8f s[2];
#pragma unroll
    for (int t = 0; t < 2; ++t) {
      const unsigned short* Krow = Kt + (long long)(j0 + t * 16 + l16) * DKV;
      v8f sc = {0.f, 0.f, 0.f, 0.f, 0.f, 0.f, 0.f, 0.f};
#pragma unroll
      for (int kk = 0; kk < 2; ++kk) {
        uint4 lo = *(const uint4*)(Krow + kk * 32 + lh * 8);
        uint4 hi = *(const uint4*)(Krow + kk * 32 + 16 + lh * 8);
        sc = __builtin_amdgcn_wmma_f32_16x16x32_bf16(
            false, aq[kk], false, pack_bf16(lo, hi), (short)0, sc, false, false);
      }
      s[t] = sc;
    }

    // ---- bias + mask + online softmax (C layout: row m = lh*8+v across 16 lanes)
    float p0[8], p1[8];
#pragma unroll
    for (int v = 0; v < 8; ++v) {
      int q  = q0 + lh * 8 + v;
      int k0 = j0 + l16;
      int k1 = k0 + 16;
      float s0 = s[0][v] + brow[k0 - q] + am[(long long)q * SEQ + k0];
      float s1 = s[1][v] + brow[k1 - q] + am[(long long)q * SEQ + k1];
      float mx = fmaxf(s0, s1);
#pragma unroll
      for (int msk = 1; msk < 16; msk <<= 1)
        mx = fmaxf(mx, __shfl_xor(mx, msk, 32));
      float mnew = fmaxf(rmax[v], mx);
      float corr = __expf((rmax[v] - mnew) * inv);
      float e0 = __expf((s0 - mnew) * inv);
      float e1 = __expf((s1 - mnew) * inv);
      float ps = e0 + e1;
#pragma unroll
      for (int msk = 1; msk < 16; msk <<= 1)
        ps += __shfl_xor(ps, msk, 32);
      rsum[v] = rsum[v] * corr + ps;
      rmax[v] = mnew;
      o[0][v] *= corr; o[1][v] *= corr; o[2][v] *= corr; o[3][v] *= corr;
      p0[v] = e0; p1[v] = e1;
    }

    // ---- relayout P (C layout -> A fragment) through the per-wave LDS tile.
    // DS ops are in-order within a wave; s_wait_dscnt is a hard scheduling fence.
#pragma unroll
    for (int v = 0; v < 8; ++v) {
      int m = lh * 8 + v;
      pl[m * 32 + l16]      = f2bf(p0[v]);
      pl[m * 32 + 16 + l16] = f2bf(p1[v]);
    }
    asm volatile("s_wait_dscnt 0x0" ::: "memory");
    uint4 plo = *(const uint4*)(pl + l16 * 32 + lh * 8);
    uint4 phi = *(const uint4*)(pl + l16 * 32 + 16 + lh * 8);
    v16bf pa = pack_bf16(plo, phi);

    // ---- O += P(16x32) * V(32x64): B-fragments direct from transposed V
#pragma unroll
    for (int c = 0; c < 4; ++c) {
      const unsigned short* vr = Vt + (long long)(c * 16 + l16) * SEQ + j0 + lh * 8;
      uint4 lo = *(const uint4*)(vr);
      uint4 hi = *(const uint4*)(vr + 16);
      o[c] = __builtin_amdgcn_wmma_f32_16x16x32_bf16(
          false, pa, false, pack_bf16(lo, hi), (short)0, o[c], false, false);
    }
  }

  // ---- epilogue: normalize, store bf16 context in [B,S,H*dk]
#pragma unroll
  for (int v = 0; v < 8; ++v) {
    float invl = 1.0f / rsum[v];
    int q = q0 + lh * 8 + v;
    unsigned short* orow = ctx + ((long long)b * SEQ + q) * DM + h * DKV + l16;
    orow[0]  = f2bf(o[0][v] * invl);
    orow[16] = f2bf(o[1][v] * invl);
    orow[32] = f2bf(o[2][v] * invl);
    orow[48] = f2bf(o[3][v] * invl);
  }
}

// ---------------------------------------------------------------- out projection
__global__ void __launch_bounds__(32)
gemm_out_kernel(const unsigned short* __restrict__ Ab,
                const unsigned short* __restrict__ Wb,
                const float* __restrict__ bias,
                float* __restrict__ out) {
  int lane = threadIdx.x;
  int lh   = lane >> 4;
  int l16  = lane & 15;
  int row0 = blockIdx.x * 32;
  int col0 = blockIdx.y * 64;

  v8f acc[2][4];
#pragma unroll
  for (int mt = 0; mt < 2; ++mt)
#pragma unroll
    for (int nt = 0; nt < 4; ++nt)
      acc[mt][nt] = (v8f){0.f,0.f,0.f,0.f,0.f,0.f,0.f,0.f};

  gemm_tile_2x4(Ab + (long long)(row0 + l16) * DM,
                Wb + (long long)(col0 + l16) * DM, lh, acc);

  int mbase = lh * 8;
#pragma unroll
  for (int nt = 0; nt < 4; ++nt) {
    int c = col0 + nt * 16 + l16;
    float bcol = bias[c];
#pragma unroll
    for (int mt = 0; mt < 2; ++mt)
#pragma unroll
      for (int v = 0; v < 8; ++v)
        out[(long long)(row0 + mt * 16 + mbase + v) * DM + c] =
            acc[mt][nt][v] + bcol;
  }
}

// ---------------------------------------------------------------- launch
extern "C" void kernel_launch(void* const* d_in, const int* in_sizes, int n_in,
                              void* d_out, int out_size, void* d_ws, size_t ws_size,
                              hipStream_t stream) {
  (void)in_sizes; (void)n_in; (void)out_size; (void)ws_size;
  const float* hidden  = (const float*)d_in[0];
  const float* amask   = (const float*)d_in[1];
  const float* Wq = (const float*)d_in[2];  const float* bq = (const float*)d_in[3];
  const float* Wk = (const float*)d_in[4];  const float* bk = (const float*)d_in[5];
  const float* Wv = (const float*)d_in[6];  const float* bv = (const float*)d_in[7];
  const float* Wo = (const float*)d_in[8];  const float* bo = (const float*)d_in[9];
  const float* rel_emb = (const float*)d_in[10];

  char* ws = (char*)d_ws;
  size_t off = 0;
  auto alloc = [&](size_t bytes) -> void* {
    void* p = ws + off;
    off = (off + bytes + 255) & ~(size_t)255;
    return p;
  };
  unsigned short* Xb   = (unsigned short*)alloc((size_t)MROWS * DM * 2);
  unsigned short* Wqb  = (unsigned short*)alloc((size_t)DM * DM * 2);
  unsigned short* Wkb  = (unsigned short*)alloc((size_t)DM * DM * 2);
  unsigned short* Wvb  = (unsigned short*)alloc((size_t)DM * DM * 2);
  unsigned short* Wob  = (unsigned short*)alloc((size_t)DM * DM * 2);
  unsigned short* Qb   = (unsigned short*)alloc((size_t)MROWS * DM * 2);
  unsigned short* Kbuf = (unsigned short*)alloc((size_t)MROWS * DM * 2);
  unsigned short* Vbuf = (unsigned short*)alloc((size_t)MROWS * DM * 2);  // [B,H,dk,S]
  unsigned short* Ctx  = (unsigned short*)alloc((size_t)MROWS * DM * 2);
  float* bias_full     = (float*)alloc((size_t)NH * RELSPAN * 4);

  float* out_proj = (float*)d_out;
  float* masks    = (float*)d_out + (size_t)MROWS * DM;

  const int nX4 = MROWS * DM / 4;
  const int nW4 = DM * DM / 4;
  cvt_bf16_kernel<<<(nX4 + 255) / 256, 256, 0, stream>>>(hidden, Xb, nX4);
  cvt_bf16_kernel<<<(nW4 + 255) / 256, 256, 0, stream>>>(Wq, Wqb, nW4);
  cvt_bf16_kernel<<<(nW4 + 255) / 256, 256, 0, stream>>>(Wk, Wkb, nW4);
  cvt_bf16_kernel<<<(nW4 + 255) / 256, 256, 0, stream>>>(Wv, Wvb, nW4);
  cvt_bf16_kernel<<<(nW4 + 255) / 256, 256, 0, stream>>>(Wo, Wob, nW4);

  bias_lut_kernel<<<(NH * RELSPAN + 255) / 256, 256, 0, stream>>>(rel_emb, bias_full);

  masks_kernel<<<dim3(BATCH * NH * SEQ), 256, 0, stream>>>(bias_full, amask, masks);

  gemm_qkv_kernel<<<dim3(MROWS / 32, DM / 64, 3), 32, 0, stream>>>(
      Xb, Wqb, Wkb, Wvb, bq, bk, bv, Qb, Kbuf, Vbuf);

  flash_attn_kernel<<<dim3(SEQ / 64, NH, BATCH), 128, 0, stream>>>(
      Qb, Kbuf, Vbuf, bias_full, amask, Ctx);

  gemm_out_kernel<<<dim3(MROWS / 32, DM / 64), 32, 0, stream>>>(
      Ctx, Wob, bo, out_proj);
}
